// Attn_29609504539278
// MI455X (gfx1250) — matrix operand order repaired and checked by reading, not compiled, not run on hardware
//
#include <hip/hip_runtime.h>
#include <hip/hip_bf16.h>

typedef __attribute__((ext_vector_type(16))) __bf16 v16bf;
typedef __attribute__((ext_vector_type(8)))  float  v8f;

union Frag { unsigned int u[8]; v16bf v; };

__device__ __forceinline__ unsigned short f2bf(float f) {
  unsigned int u = __builtin_bit_cast(unsigned int, f);
  u += 0x7FFFu + ((u >> 16) & 1u);            // round-to-nearest-even
  return (unsigned short)(u >> 16);
}

// Low 32 bits of a generic pointer to __shared__ == LDS byte offset (ISA 10.2).
__device__ __forceinline__ unsigned lds_off(const void* p) {
  return (unsigned)(unsigned long long)p;
}

// CDNA5 async global->LDS copy, 16B per lane, tracked by ASYNCcnt.
__device__ __forceinline__ void async_b128(unsigned lds, const void* g) {
  asm volatile("global_load_async_to_lds_b128 %0, %1, off"
               :: "v"(lds), "v"(g) : "memory");
}
template <int N>
__device__ __forceinline__ void wait_async() {
  asm volatile("s_wait_asynccnt %0" :: "n"(N) : "memory");
}

// ---------------------------------------------------------------- convert ---
__global__ __launch_bounds__(256) void cvt_f32_bf16(const float* __restrict__ in,
                                                    unsigned short* __restrict__ out,
                                                    int n) {
  int i = blockIdx.x * 256 + threadIdx.x;
  if (i < n) out[i] = f2bf(in[i]);
}

// fp32 [K=1024][N=1024] -> bf16 transposed [N][K] (once; GEMM B tiles then
// stage with contiguous async copies, no in-kernel transpose).
__global__ __launch_bounds__(256) void cvt_w_bf16_t(const float* __restrict__ in,
                                                    unsigned short* __restrict__ out) {
  int i = blockIdx.x * 256 + threadIdx.x;     // grid covers 1024*1024
  int k = i >> 10, n = i & 1023;
  out[(size_t)n * 1024 + k] = f2bf(in[i]);
}

// ------------------------------------------------------------------- GEMM ---
// C[4096][1024] = A[4096][1024](bf16) * Bt^T  (Bt is [N][K] bf16) + bias
// mode 1: bf16 head-major [B,H,S,64]; mode 2: fp32 row-major;
// mode 3: bf16 transposed head-major [B,H,64,S] (for V).
__global__ __launch_bounds__(256) void gemm_bf16(const unsigned short* __restrict__ A,
                                                 const unsigned short* __restrict__ Bt,
                                                 const float* __restrict__ bias,
                                                 void* __restrict__ out, int mode) {
  __shared__ unsigned short lA[2][256 * 32];   // double-buffered A tile  2x16KB
  __shared__ unsigned short lBt[2][64 * 32];   // double-buffered B tile  2x 4KB
  const int tid  = threadIdx.x;
  const int m0   = blockIdx.y * 256;
  const int n0   = blockIdx.x * 64;
  const int wave = tid >> 5, lane = tid & 31, ln = lane & 15, half = lane >> 4;
  const unsigned lA_off  = lds_off(&lA[0][0]);
  const unsigned lBt_off = lds_off(&lBt[0][0]);
  const char* Ab  = (const char*)A;            // row stride 2048 B
  const char* Btb = (const char*)Bt;           // row stride 2048 B

  // Stage one 256x32 A tile + 64x32 B tile: 5 async b128 per thread.
  auto stage = [&](int buf, int kt) {
    unsigned ao = lA_off + (unsigned)buf * 16384u;
    unsigned bo = lBt_off + (unsigned)buf * 4096u;
    #pragma unroll
    for (int it = 0; it < 4; ++it) {           // A: 1024 x 16B chunks
      int c = tid + 256 * it;
      int row = c >> 2, q = c & 3;             // 4 chunks per 64B row
      async_b128(ao + c * 16, Ab + (size_t)(m0 + row) * 2048 + kt * 64 + q * 16);
    }
    int c = tid, n = c >> 2, q = c & 3;        // B: 256 x 16B chunks
    async_b128(bo + c * 16, Btb + (size_t)(n0 + n) * 2048 + kt * 64 + q * 16);
  };

  v8f acc[2][4];
  #pragma unroll
  for (int ms = 0; ms < 2; ++ms)
    #pragma unroll
    for (int t = 0; t < 4; ++t)
      #pragma unroll
      for (int r = 0; r < 8; ++r) acc[ms][t][r] = 0.f;

  stage(0, 0);
  for (int kt = 0; kt < 32; ++kt) {            // K = 1024 in steps of 32
    const int buf = kt & 1;
    if (kt + 1 < 32) {                         // overlap next DMA with compute
      stage(buf ^ 1, kt + 1);
      wait_async<5>();                         // in-order: tile kt has landed
    } else {
      wait_async<0>();
    }
    __syncthreads();

    const unsigned int* lAd  = (const unsigned int*)&lA[buf][0];
    const unsigned int* lBtd = (const unsigned int*)&lBt[buf][0];
    Frag af[2];
    #pragma unroll
    for (int ms = 0; ms < 2; ++ms) {
      int mrow = wave * 32 + ms * 16 + ln;
      #pragma unroll
      for (int d = 0; d < 8; ++d) {            // A frag K-pair dwords (contiguous k = 16*half + s)
        int dw = d + (half ? 8 : 0);
        af[ms].u[d] = lAd[mrow * 16 + dw];
      }
    }
    #pragma unroll
    for (int t = 0; t < 4; ++t) {
      Frag bf_;
      int nr = t * 16 + ln;
      #pragma unroll
      for (int d = 0; d < 8; ++d) bf_.u[d] = lBtd[nr * 16 + d + (half ? 8 : 0)];
      #pragma unroll
      for (int ms = 0; ms < 2; ++ms)
        acc[ms][t] = __builtin_amdgcn_wmma_f32_16x16x32_bf16(false, af[ms].v, false, bf_.v,
                                                             (short)0, acc[ms][t], false, false);
    }
    __syncthreads();                           // readers done before overwrite
  }

  #pragma unroll
  for (int ms = 0; ms < 2; ++ms)
    #pragma unroll
    for (int t = 0; t < 4; ++t) {
      int col = n0 + t * 16 + ln;
      float bvv = bias ? bias[col] : 0.f;
      #pragma unroll
      for (int r = 0; r < 8; ++r) {
        int row = m0 + wave * 32 + ms * 16 + half * 8 + r;
        float vv = acc[ms][t][r] + bvv;
        if (mode == 2) {
          ((float*)out)[(size_t)row * 1024 + col] = vv;
        } else {
          int b = row >> 11, s = row & 2047, h = col >> 6, dd = col & 63;
          if (mode == 1)
            ((unsigned short*)out)[((size_t)(b * 16 + h) * 2048 + s) * 64 + dd] = f2bf(vv);
          else  // mode 3: V^T head-major [B,H,64,S]
            ((unsigned short*)out)[((size_t)(b * 16 + h) * 64 + dd) * 2048 + s] = f2bf(vv);
        }
      }
    }
}

// -------------------------------------------------------- flash attention ---
// grid: (q_tile 0..31, bh 0..31), 128 threads = 4 waves x 16 q-rows.
// Q,K head-major [B,H,S,64]; Vt transposed head-major [B,H,64,S].
__global__ __launch_bounds__(128) void flash_attn(const unsigned short* __restrict__ Q,
                                                  const unsigned short* __restrict__ K,
                                                  const unsigned short* __restrict__ Vt,
                                                  unsigned short* __restrict__ O) {
  __shared__ unsigned short lK[2][64 * 64];    // K tile [key][64 dk]   2x8KB
  __shared__ unsigned short lVt[2][64 * 64];   // V^T tile [dv][key]    2x8KB
  __shared__ unsigned short lP[4][16 * 64];    // per-wave P [m][key]     8KB
  const int tid  = threadIdx.x;
  const int wave = tid >> 5, lane = tid & 31, ln = lane & 15, half = lane >> 4;
  const int bh   = blockIdx.y;
  const int q0   = blockIdx.x * 64;
  const int njt  = (int)blockIdx.x;            // last key-tile index (diagonal)
  const char* Kb  = (const char*)K  + (size_t)bh * 2048 * 128;  // key row = 128B
  const char* Vtb = (const char*)Vt + (size_t)bh * 64 * 4096;   // dv row = 4096B
  const unsigned int* Qd = (const unsigned int*)Q + (size_t)bh * 2048 * 32;
  const unsigned lK_off  = lds_off(&lK[0][0]);
  const unsigned lVt_off = lds_off(&lVt[0][0]);

  // Stage one K tile + one V^T tile: 8 async b128 per thread.
  auto stage = [&](int buf, int jt) {
    int j0 = jt * 64;
    unsigned ko = lK_off  + (unsigned)buf * 8192u;
    unsigned vo = lVt_off + (unsigned)buf * 8192u;
    #pragma unroll
    for (int it = 0; it < 4; ++it) {           // 512 x 16B chunks per tile
      int c = tid + 128 * it;
      int row = c >> 3, q = c & 7;             // 8 chunks per 128B row
      async_b128(ko + c * 16, Kb  + (size_t)(j0 + row) * 128 + q * 16);
      async_b128(vo + c * 16, Vtb + (size_t)row * 4096 + j0 * 2 + q * 16);
    }
  };

  Frag qf[2];                                   // Q A-frags, two dk steps of 32
  #pragma unroll
  for (int ks = 0; ks < 2; ++ks)
    #pragma unroll
    for (int d = 0; d < 8; ++d) {
      int dw = ks * 16 + d + (half ? 8 : 0);
      qf[ks].u[d] = Qd[(size_t)(q0 + wave * 16 + ln) * 32 + dw];
    }

  float mrow[8], lrow[8];
  v8f o[4];
  #pragma unroll
  for (int r = 0; r < 8; ++r) { mrow[r] = -__builtin_inff(); lrow[r] = 0.f; }
  #pragma unroll
  for (int t = 0; t < 4; ++t)
    #pragma unroll
    for (int r = 0; r < 8; ++r) o[t][r] = 0.f;

  const float SCL = 0.125f * 1.44269504088896f;   // 1/sqrt(64) * log2(e)

  stage(0, 0);
  for (int jt = 0; jt <= njt; ++jt) {
    const int j0  = jt * 64;
    const int buf = jt & 1;
    if (jt + 1 <= njt) {                       // overlap next DMA with compute
      stage(buf ^ 1, jt + 1);
      wait_async<8>();                         // in-order: tile jt has landed
    } else {
      wait_async<0>();
    }
    __syncthreads();
    if (jt + 2 <= njt)                         // warm L2 for tile jt+2
      __builtin_prefetch(Kb + (size_t)(j0 + 128) * 128, 0, 0);

    v8f s[4];                                  // S = Q K^T (8 wmma)
    #pragma unroll
    for (int t = 0; t < 4; ++t)
      #pragma unroll
      for (int r = 0; r < 8; ++r) s[t][r] = 0.f;
    const unsigned int* lKd = (const unsigned int*)&lK[buf][0];
    #pragma unroll
    for (int ks = 0; ks < 2; ++ks)
      #pragma unroll
      for (int t = 0; t < 4; ++t) {
        Frag bf_;
        int kr = t * 16 + ln;
        #pragma unroll
        for (int d = 0; d < 8; ++d)
          bf_.u[d] = lKd[kr * 32 + ks * 16 + d + (half ? 8 : 0)];
        s[t] = __builtin_amdgcn_wmma_f32_16x16x32_bf16(false, qf[ks].v, false, bf_.v,
                                                       (short)0, s[t], false, false);
      }

    const bool diag = (jt == njt);             // online softmax (fp32)
    #pragma unroll
    for (int r = 0; r < 8; ++r) {
      int qrow = q0 + wave * 16 + half * 8 + r;
      float fs[4];
      #pragma unroll
      for (int t = 0; t < 4; ++t) {
        float sv = s[t][r] * SCL;
        if (diag && (j0 + t * 16 + ln) > qrow) sv = -__builtin_inff();
        fs[t] = sv;
      }
      float tmax = fmaxf(fmaxf(fs[0], fs[1]), fmaxf(fs[2], fs[3]));
      #pragma unroll
      for (int msk = 1; msk <= 8; msk <<= 1)
        tmax = fmaxf(tmax, __shfl_xor(tmax, msk, 32));
      float mn   = fmaxf(mrow[r], tmax);
      float corr = exp2f(mrow[r] - mn);
      float psum = 0.f;
      #pragma unroll
      for (int t = 0; t < 4; ++t) {
        float p = exp2f(fs[t] - mn);
        psum += p;
        lP[wave][(half * 8 + r) * 64 + t * 16 + ln] = f2bf(p);
      }
      #pragma unroll
      for (int msk = 1; msk <= 8; msk <<= 1) psum += __shfl_xor(psum, msk, 32);
      lrow[r] = lrow[r] * corr + psum;
      mrow[r] = mn;
      #pragma unroll
      for (int t = 0; t < 4; ++t) o[t][r] *= corr;
    }
    asm volatile("s_wait_dscnt 0x0" ::: "memory");  // wave-local P RAW via LDS

    const unsigned int* lPd = (const unsigned int*)(&lP[wave][0]);
    const unsigned int* lVd = (const unsigned int*)&lVt[buf][0];
    #pragma unroll
    for (int ks = 0; ks < 2; ++ks) {           // O += P V (8 wmma)
      Frag pf;
      #pragma unroll
      for (int d = 0; d < 8; ++d)
        pf.u[d] = lPd[ln * 32 + ks * 16 + d + (half ? 8 : 0)];
      #pragma unroll
      for (int t = 0; t < 4; ++t) {
        Frag vf;
        int dvr = t * 16 + ln;
        #pragma unroll
        for (int d = 0; d < 8; ++d)
          vf.u[d] = lVd[dvr * 32 + ks * 16 + d + (half ? 8 : 0)];
        o[t] = __builtin_amdgcn_wmma_f32_16x16x32_bf16(false, pf.v, false, vf.v,
                                                       (short)0, o[t], false, false);
      }
    }
    __syncthreads();                           // readers done before overwrite
  }

  const int b = bh >> 4, h = bh & 15;          // write attn out [B,S,H*64] bf16
  #pragma unroll
  for (int r = 0; r < 8; ++r) {
    int srow = q0 + wave * 16 + half * 8 + r;
    float inv = 1.f / lrow[r];
    #pragma unroll
    for (int t = 0; t < 4; ++t)
      O[((size_t)(b * 2048 + srow)) * 1024 + h * 64 + t * 16 + ln] = f2bf(o[t][r] * inv);
  }
}

// ------------------------------------------------------------------ launch ---
extern "C" void kernel_launch(void* const* d_in, const int* in_sizes, int n_in,
                              void* d_out, int out_size, void* d_ws, size_t ws_size,
                              hipStream_t stream) {
  (void)in_sizes; (void)n_in; (void)out_size; (void)ws_size;
  const float* x  = (const float*)d_in[0];
  const float* Wq = (const float*)d_in[1];
  const float* bq = (const float*)d_in[2];
  const float* Wk = (const float*)d_in[3];
  const float* bk = (const float*)d_in[4];
  const float* Wv = (const float*)d_in[5];
  const float* bv = (const float*)d_in[6];
  const float* Wo = (const float*)d_in[7];
  float* out = (float*)d_out;

  char* ws = (char*)d_ws;
  const size_t MB = 1024 * 1024;
  unsigned short* xb  = (unsigned short*)(ws + 0);        // x bf16            8 MiB
  unsigned short* wqT = (unsigned short*)(ws + 8  * MB);  // Wq^T bf16 [N][K]  2 MiB
  unsigned short* wkT = (unsigned short*)(ws + 10 * MB);
  unsigned short* wvT = (unsigned short*)(ws + 12 * MB);
  unsigned short* woT = (unsigned short*)(ws + 14 * MB);
  unsigned short* Qh  = (unsigned short*)(ws + 16 * MB);  // [B,H,S,64]        8 MiB
  unsigned short* Kh  = (unsigned short*)(ws + 24 * MB);
  unsigned short* VhT = (unsigned short*)(ws + 32 * MB);  // [B,H,64,S]        8 MiB
  unsigned short* Ah  = (unsigned short*)(ws + 40 * MB);  // attn out bf16     8 MiB

  const int nx = 4096 * 1024, nw = 1024 * 1024;
  cvt_f32_bf16<<<dim3((nx + 255) / 256), dim3(256), 0, stream>>>(x, xb, nx);
  cvt_w_bf16_t<<<dim3(nw / 256), dim3(256), 0, stream>>>(Wq, wqT);
  cvt_w_bf16_t<<<dim3(nw / 256), dim3(256), 0, stream>>>(Wk, wkT);
  cvt_w_bf16_t<<<dim3(nw / 256), dim3(256), 0, stream>>>(Wv, wvT);
  cvt_w_bf16_t<<<dim3(nw / 256), dim3(256), 0, stream>>>(Wo, woT);

  dim3 gg(16, 16), gb(256);                    // N/64 x M/256
  gemm_bf16<<<gg, gb, 0, stream>>>(xb, wqT, bq, Qh, 1);
  gemm_bf16<<<gg, gb, 0, stream>>>(xb, wkT, bk, Kh, 1);
  gemm_bf16<<<gg, gb, 0, stream>>>(xb, wvT, bv, VhT, 3);

  flash_attn<<<dim3(32, 32), dim3(128), 0, stream>>>(Qh, Kh, VhT, Ah);

  gemm_bf16<<<gg, gb, 0, stream>>>(Ah, woT, nullptr, out, 2);
}
